// W8A16Linear_41781441855910
// MI455X (gfx1250) — matrix operand order repaired and checked by reading, not compiled, not run on hardware
//
#include <hip/hip_runtime.h>
#include <stdint.h>

// W8A16 GEMM: out[B,S,OUT] = A[B,S,IN] x dequant(Wq[IN,OUT]) + bias
// M = B*S = 8192, K = IN = 4096, N = OUT = 16384. Compute-bound (AI ~2750 F/B)
// -> fp16 WMMA (v_wmma_f32_16x16x32_f16). A tile is DMA'd into LDS by the
// CDNA5 Tensor Data Mover (TENSOR_LOAD_TO_LDS, TENSORcnt), one instruction
// per K-step issued by wave 0. int8 weights are dequantized in-register into
// a transposed [n][k] LDS tile so B fragments are straight ds_load_b128s.

typedef __attribute__((ext_vector_type(16))) _Float16 v16h;
typedef __attribute__((ext_vector_type(8)))  _Float16 v8h;
typedef __attribute__((ext_vector_type(2)))  _Float16 v2h;
typedef __attribute__((ext_vector_type(8)))  float    v8f;
typedef __attribute__((ext_vector_type(4)))  unsigned v4u;
typedef __attribute__((ext_vector_type(8)))  unsigned v8u;

#define M_DIM (4 * 2048)
#define N_DIM 16384
#define K_DIM 4096
#define BLK_M 128
#define BLK_N 128
#define BLK_K 32
#define NKT   (K_DIM / BLK_K)   // 128 K-steps

union FragH {
  v16h f;
  v8h  h[2];
};

__device__ __forceinline__ int sext8(uint2 w, int j) {
  unsigned r = (j < 4) ? w.x : w.y;
  return (int)(signed char)(r >> ((j & 3) * 8));
}

// ---- Tensor Data Mover: one D#-driven DMA of a [tile_dim1 x tile_dim0]
// sub-tile of a 2D row-major tensor into LDS (contiguous, X-then-Y order).
// D# group0: count=1 | lds_addr | global_addr[56:0] | type=2 ("image").
// D# group1: data_size=2B | tensor_dim0/1 | tile_dim0/1 | tensor_dim0_stride.
__device__ __forceinline__ void tdm_load_tile_2d(uint32_t lds_addr, uint64_t gaddr) {
  v4u g0;
  g0[0] = 1u;                                   // count=1, user mode
  g0[1] = lds_addr;                             // LDS byte address
  g0[2] = (uint32_t)gaddr;                      // global_addr[31:0]
  g0[3] = (uint32_t)((gaddr >> 32) & 0x01FFFFFFu) | (2u << 30);  // addr[56:32] | type=2

  v8u g1;
  g1[0] = (1u << 16);                           // data_size=1 (2 bytes), no mask/flags
  g1[1] = ((uint32_t)K_DIM & 0xFFFFu) << 16;    // tensor_dim0[15:0]  @bits 63:48
  g1[2] = ((uint32_t)K_DIM >> 16)               // tensor_dim0[31:16] @bits 79:64
        | (((uint32_t)M_DIM & 0xFFFFu) << 16);  // tensor_dim1[15:0]  @bits 95:80
  g1[3] = ((uint32_t)M_DIM >> 16)               // tensor_dim1[31:16] @bits 111:96
        | ((uint32_t)BLK_K << 16);              // tile_dim0=32       @bits 127:112
  g1[4] = (uint32_t)BLK_M;                      // tile_dim1=128, tile_dim2=0
  g1[5] = (uint32_t)K_DIM;                      // tensor_dim0_stride[31:0]
  g1[6] = 0u;                                   // stride hi, dim1_stride lo
  g1[7] = 0u;

  // 2D tensor: VADDR2/VADDR3 = NULL -> 2-operand form.
  asm volatile("tensor_load_to_lds %0, %1" :: "s"(g0), "s"(g1) : "memory");
}

__device__ __forceinline__ void wait_tensorcnt0() {
#if __has_builtin(__builtin_amdgcn_s_wait_tensorcnt)
  __builtin_amdgcn_s_wait_tensorcnt(0);
#else
  asm volatile("s_wait_tensorcnt 0" ::: "memory");
#endif
}

__global__ __launch_bounds__(256)
void W8A16Linear_41781441855910_kernel(const _Float16* __restrict__ A,
                                       const int8_t*  __restrict__ Wq,
                                       const _Float16* __restrict__ scales,
                                       const _Float16* __restrict__ bias,
                                       _Float16* __restrict__ out) {
  // Double-buffered LDS: A tile [m][k] fp16 (TDM-filled), W tile [n][k] fp16
  __shared__ _Float16 ldsA[2][BLK_M * BLK_K];   // 2 x 8 KB
  __shared__ _Float16 ldsW[2][BLK_N * BLK_K];   // 2 x 8 KB

  const int tid  = threadIdx.x;
  const int lane = tid & 31;
  const int wave = tid >> 5;
  const int wm   = wave & 1;    // wave tile row (0..1) -> 64 rows each
  const int wn   = wave >> 1;   // wave tile col (0..3) -> 32 cols each

  const int mBase = blockIdx.y * BLK_M;
  const int nBase = blockIdx.x * BLK_N;

  // A-tile global base: row mBase, k advances by BLK_K halfs per step
  const uint64_t aTileBase = (uint64_t)(uintptr_t)(A + (size_t)mBase * K_DIM);
  uint32_t ldsAaddr[2];
  ldsAaddr[0] = (uint32_t)(uintptr_t)&ldsA[0][0];
  ldsAaddr[1] = (uint32_t)(uintptr_t)&ldsA[1][0];

  // ---- W-tile staging: thread covers 2 k-rows x 8 n-cols of int8 ----
  const int wK  = (tid >> 4) * 2;     // 0,2,...,30
  const int wNo = (tid & 15) * 8;     // n offset within tile, 0..120

  // per-thread dequant scales for its 8 columns (loaded once)
  const v8h sc = *(const v8h*)(scales + nBase + wNo);

  uint2 wReg0, wReg1;

  // ---------- prologue: fill buffer 0 for K-tile 0 ----------
  if (wave == 0) {
    tdm_load_tile_2d(ldsAaddr[0], aTileBase);   // TENSORcnt++
  }
  wReg0 = *(const uint2*)(Wq + (size_t)(wK)     * N_DIM + nBase + wNo);
  wReg1 = *(const uint2*)(Wq + (size_t)(wK + 1) * N_DIM + nBase + wNo);
#pragma unroll
  for (int j = 0; j < 8; ++j) {
    v2h hv; hv[0] = (_Float16)sext8(wReg0, j); hv[1] = (_Float16)sext8(wReg1, j);
    v2h sv; sv[0] = sc[j]; sv[1] = sc[j];
    hv = hv * sv;                               // v_pk_mul_f16
    *(v2h*)&ldsW[0][(wNo + j) * BLK_K + wK] = hv;
  }
  if (wave == 0) {
    wait_tensorcnt0();  // tile in LDS before wave 0 reaches the barrier
  }

  // ---------- main K loop ----------
  v8f acc[4][2] = {};

  const int l15   = lane & 15;
  const int hiH   = lane >> 4;        // 0 or 1 (lane half)
  const int koffA = hiH * 8;          // A frag: K 0-7/16-23 vs 8-15/24-31
  const int koffB = hiH * 8;          // B frag: K 0-7/16-23 vs 8-15/24-31

  for (int kt = 0; kt < NKT; ++kt) {
    const int cur = kt & 1;
    const int nxt = cur ^ 1;
    const bool has = (kt + 1) < NKT;

    __syncthreads();   // buffer `cur` fully written; buffer `nxt` fully consumed

    if (has) {
      const int k0 = (kt + 1) * BLK_K;
      // one TDM DMA moves the whole next A tile into LDS
      if (wave == 0) {
        tdm_load_tile_2d(ldsAaddr[nxt], aTileBase + (uint64_t)k0 * sizeof(_Float16));
      }
      // next weight tile into registers (needs dequant before LDS)
      wReg0 = *(const uint2*)(Wq + (size_t)(k0 + wK)     * N_DIM + nBase + wNo);
      wReg1 = *(const uint2*)(Wq + (size_t)(k0 + wK + 1) * N_DIM + nBase + wNo);
      // warm L2 with the weight stream 4 K-tiles ahead (weights fit in 192MB L2)
      if (kt + 4 < NKT)
        __builtin_prefetch(Wq + (size_t)((kt + 4) * BLK_K + wK) * N_DIM + nBase + wNo, 0, 1);
    }

    // ---- load fragments from LDS (layouts per CDNA5 ISA 7.12.2) ----
    FragH af[4], bf[2];
#pragma unroll
    for (int mt = 0; mt < 4; ++mt) {
      const int m = wm * 64 + mt * 16 + l15;
      af[mt].h[0] = *(const v8h*)&ldsA[cur][m * BLK_K + koffA];
      af[mt].h[1] = *(const v8h*)&ldsA[cur][m * BLK_K + 16 + koffA];
    }
#pragma unroll
    for (int nt = 0; nt < 2; ++nt) {
      const int n = wn * 32 + nt * 16 + l15;
      bf[nt].h[0] = *(const v8h*)&ldsW[cur][n * BLK_K + koffB];
      bf[nt].h[1] = *(const v8h*)&ldsW[cur][n * BLK_K + koffB + 16];
    }

    // ---- 8 WMMAs: 64x32 wave tile ----
#pragma unroll
    for (int mt = 0; mt < 4; ++mt)
#pragma unroll
      for (int nt = 0; nt < 2; ++nt)
        acc[mt][nt] = __builtin_amdgcn_wmma_f32_16x16x32_f16(
            false, af[mt].f, false, bf[nt].f,
            (short)0, acc[mt][nt], false, false);

    // ---- dequant + store next weight tile into the other buffer ----
    if (has) {
#pragma unroll
      for (int j = 0; j < 8; ++j) {
        v2h hv; hv[0] = (_Float16)sext8(wReg0, j); hv[1] = (_Float16)sext8(wReg1, j);
        v2h sv; sv[0] = sc[j]; sv[1] = sc[j];
        hv = hv * sv;                           // v_pk_mul_f16
        *(v2h*)&ldsW[nxt][(wNo + j) * BLK_K + wK] = hv;
      }
    }

    // wave 0's TDM must complete before it arrives at the next barrier
    if (wave == 0) {
      wait_tensorcnt0();
    }
  }

  // ---------- epilogue: bias add, f32 -> f16, store ----------
  const int m0 = mBase + wm * 64;
  const int n0 = nBase + wn * 32;

  float bv[2];
#pragma unroll
  for (int nt = 0; nt < 2; ++nt)
    bv[nt] = (float)bias[n0 + nt * 16 + l15];

#pragma unroll
  for (int mt = 0; mt < 4; ++mt) {
#pragma unroll
    for (int nt = 0; nt < 2; ++nt) {
      const int col = n0 + nt * 16 + l15;
#pragma unroll
      for (int r = 0; r < 8; ++r) {
        // C/D layout: VGPR r holds M=r (lanes 0-15) and M=r+8 (lanes 16-31)
        const int row = m0 + mt * 16 + r + hiH * 8;
        out[(size_t)row * N_DIM + col] = (_Float16)(acc[mt][nt][r] + bv[nt]);
      }
    }
  }
}

extern "C" void kernel_launch(void* const* d_in, const int* in_sizes, int n_in,
                              void* d_out, int out_size, void* d_ws, size_t ws_size,
                              hipStream_t stream) {
  (void)in_sizes; (void)n_in; (void)out_size; (void)d_ws; (void)ws_size;
  const _Float16* A      = (const _Float16*)d_in[0];
  const int8_t*   Wq     = (const int8_t*)d_in[1];
  const _Float16* scales = (const _Float16*)d_in[2];
  const _Float16* bias   = (const _Float16*)d_in[3];
  _Float16*       out    = (_Float16*)d_out;

  dim3 grid(N_DIM / BLK_N, M_DIM / BLK_M);   // 128 x 64 blocks
  dim3 block(256);
  W8A16Linear_41781441855910_kernel<<<grid, block, 0, stream>>>(A, Wq, scales, bias, out);
}
